// EfficientFlashAttnVarlenQKVPacked_30726196035788
// MI455X (gfx1250) — matrix operand order repaired and checked by reading, not compiled, not run on hardware
//
#include <hip/hip_runtime.h>

// ---------------------------------------------------------------------------
// Types for CDNA5 WMMA fragments (wave32) + TDM descriptor groups
// ---------------------------------------------------------------------------
typedef __attribute__((ext_vector_type(16))) _Float16 v16h;
typedef __attribute__((ext_vector_type(8)))  float    v8f;
typedef __attribute__((ext_vector_type(2)))  float    v2f;
typedef __attribute__((ext_vector_type(4)))  unsigned u32x4;
typedef __attribute__((ext_vector_type(8)))  int      i32x8;
typedef __attribute__((ext_vector_type(4)))  int      i32x4;

#define T_TOK 4096
#define NH    12
#define DH    64
#define HD    768                 // NH*DH, contiguous per (token, qkv-slot)
#define QKV_STRIDE 2304           // 3*HD
#define NELEM (T_TOK*HD)
#define RANK  16

// D = A(16xK) * B(Kx16) + C, f16 inputs, f32 accum  (probe-confirmed builtin)
__device__ __forceinline__ v8f wmma_f16(v16h a, v16h b, v8f c) {
  return __builtin_amdgcn_wmma_f32_16x16x32_f16(false, a, false, b, (short)0, c, false, false);
}
// D = A(16x4) * B(4x16) + C, all f32 (ISA V_WMMA_F32_16X16X4_F32)
__device__ __forceinline__ v8f wmma_f32x4(v2f a, v2f b, v8f c) {
  return __builtin_amdgcn_wmma_f32_16x16x4_f32(false, a, false, b, (short)0, c, false, false);
}

// ---------------------------------------------------------------------------
// TDM: DMA one K/V tile (32 tokens x {K,V} x 64 f32) from global into LDS.
// 3D descriptor: dim0 = d (64 elems), dim1 = K/V slab (2, stride 768 elems),
// dim2 = token (32, stride 2304 elems). LDS result: [tok][slab][d] contiguous.
// Descriptor packing per CDNA5 ISA.
// ---------------------------------------------------------------------------
#if __has_builtin(__builtin_amdgcn_tensor_load_to_lds)
#define HAVE_TDM 1
__device__ __forceinline__ void tdm_load_kv(const float* gptr, unsigned lds_off) {
  unsigned long long ga = (unsigned long long)(uintptr_t)gptr;
  u32x4 g0;
  g0[0] = 1u;                                           // count=1, no gather
  g0[1] = lds_off;                                      // lds_addr (bytes)
  g0[2] = (unsigned)(ga & 0xffffffffu);                 // global_addr[31:0]
  g0[3] = (unsigned)((ga >> 32) & 0x01ffffffu)          // global_addr[56:32]
        | 0x80000000u;                                  // type=2 ("image")
  i32x8 g1;
  g1[0] = 0x00020000;                 // wg_mask=0 | data_size=2 (4B)
  g1[1] = (int)(64u << 16);           // tensor_dim0=64  (bits 79:48, lo16)
  g1[2] = (int)(2u  << 16);           // dim0 hi=0 | tensor_dim1=2 (lo16)
  g1[3] = (int)(64u << 16);           // dim1 hi=0 | tile_dim0=64
  g1[4] = (int)(2u | (32u << 16));    // tile_dim1=2 | tile_dim2=32
  g1[5] = 768;                        // tensor_dim0_stride[31:0]
  g1[6] = (int)((2304u & 0xffffu) << 16); // stride0 hi=0 | stride1[15:0]
  g1[7] = (int)(2304u >> 16);         // tensor_dim1_stride[47:16]
  i32x4 g2; g2[0] = 32; g2[1] = 0; g2[2] = 0; g2[3] = 0;  // tensor_dim2=32
  i32x4 g3; g3[0] = 0;  g3[1] = 0; g3[2] = 0; g3[3] = 0;
#if defined(__clang_major__) && (__clang_major__ >= 23)
  i32x8 gx = {};
  __builtin_amdgcn_tensor_load_to_lds(g0, g1, g2, g3, gx, 0);
#else
  __builtin_amdgcn_tensor_load_to_lds(g0, g1, g2, g3, 0);
#endif
}
#else
#define HAVE_TDM 0
#endif

// ---------------------------------------------------------------------------
// Varlen causal flash attention, one wave32 per (16-row q tile, head)
// qkv: [T,3,H,D] f32; out: [T,H,D] f32 (first region of d_out)
// ---------------------------------------------------------------------------
__global__ __launch_bounds__(32)
void fa_varlen_kernel(const float* __restrict__ qkv, const int* __restrict__ cu,
                      float* __restrict__ out) {
  const int lane = threadIdx.x;
  const int lh   = lane & 15;
  const int half = lane >> 4;
  const int tile = blockIdx.x;
  const int h    = blockIdx.y;
  const int r0   = tile * 16;
  const float scale = 0.125f; // 1/sqrt(64)

  const int c1 = cu[1], c2 = cu[2], c3 = cu[3];
  auto segf = [&](int t) { return (t >= c1) + (t >= c2) + (t >= c3); };

  int qseg[8];
#pragma unroll
  for (int r = 0; r < 8; ++r) qseg[r] = segf(r0 + r + half * 8);

  // ---- Q A-fragments (16x32 f16, two d-chunks covering D=64) ----
  // A layout: row M = lane%16; flat elem e -> K = kbase + half*16 + e
  v16h aq0, aq1;
  {
    const float* qb = qkv + (size_t)(r0 + lh) * QKV_STRIDE + h * DH;
#pragma unroll
    for (int e = 0; e < 16; ++e) {
      int K = half * 16 + e;
      aq0[e] = (_Float16)qb[K];
      aq1[e] = (_Float16)qb[32 + K];
    }
  }

  float m_r[8], l_r[8];
  v8f acc[4] = {v8f{}, v8f{}, v8f{}, v8f{}};
#pragma unroll
  for (int r = 0; r < 8; ++r) { m_r[r] = -3.0e38f; l_r[r] = 0.f; }

  const int kend   = r0 + 15;
  const int kstart = cu[segf(r0)];

  __shared__ float    KV[32 * 128];   // [tok][ K(64) | V(64) ] f32, 16 KB
  __shared__ _Float16 Plds[16][32];

#if HAVE_TDM
  const unsigned kv_off = (unsigned)(uintptr_t)(&KV[0]);
#endif

  for (int kb = (kstart & ~31); kb <= kend; kb += 32) {
    if (kb + 32 <= kend)
      __builtin_prefetch(qkv + (size_t)(kb + 32) * QKV_STRIDE + HD + h * DH, 0, 1);

#if HAVE_TDM
    // ---- DMA K/V tile into LDS via Tensor Data Mover ----
    tdm_load_kv(qkv + (size_t)kb * QKV_STRIDE + HD + h * DH, kv_off);
    __builtin_amdgcn_s_wait_tensorcnt(0);
    __syncthreads();
#else
    for (int i = lane; i < 32 * 64; i += 32) {
      int tok = i >> 6, d = i & 63;
      KV[tok * 128 + d]      = qkv[(size_t)(kb + tok) * QKV_STRIDE + HD     + h * DH + d];
      KV[tok * 128 + 64 + d] = qkv[(size_t)(kb + tok) * QKV_STRIDE + 2 * HD + h * DH + d];
    }
    __syncthreads();
#endif

    // ---- S = Q K^T for 32 keys (two 16-key groups) ----
    int ktg[2], ksg[2];
    v8f s[2];
#pragma unroll
    for (int g = 0; g < 2; ++g) {
      const int ktl = g * 16 + lh;           // this lane's key column (local)
      const int kt  = kb + ktl;
      ktg[g] = kt; ksg[g] = segf(kt);
      const float* kl = &KV[ktl * 128];
      v16h b0, b1;
#pragma unroll
      for (int e = 0; e < 16; ++e) {         // B layout: K = half*16 + e
        int d = half * 16 + e;
        b0[e] = (_Float16)kl[d];
        b1[e] = (_Float16)kl[32 + d];
      }
      v8f cf = {};
      cf = wmma_f16(aq0, b0, cf);
      cf = wmma_f16(aq1, b1, cf);
      s[g] = cf;
    }

    // ---- online softmax (rows striped across 16-lane halves) ----
    float newm[8], sv[2][8];
#pragma unroll
    for (int r = 0; r < 8; ++r) newm[r] = m_r[r];
#pragma unroll
    for (int g = 0; g < 2; ++g)
#pragma unroll
      for (int r = 0; r < 8; ++r) {
        int qt = r0 + r + half * 8;
        bool valid = (ksg[g] == qseg[r]) && (ktg[g] <= qt);
        float x = valid ? s[g][r] * scale : -3.0e38f;
        sv[g][r] = x;
        newm[r] = fmaxf(newm[r], x);
      }
#pragma unroll
    for (int off = 1; off < 16; off <<= 1)
#pragma unroll
      for (int r = 0; r < 8; ++r)
        newm[r] = fmaxf(newm[r], __shfl_xor(newm[r], off, 32));

    float alpha[8], rs[8], p[2][8];
#pragma unroll
    for (int r = 0; r < 8; ++r) {
      alpha[r] = __expf(m_r[r] - newm[r]);
      m_r[r] = newm[r];
      rs[r] = 0.f;
    }
#pragma unroll
    for (int g = 0; g < 2; ++g)
#pragma unroll
      for (int r = 0; r < 8; ++r) {
        float pv = (sv[g][r] > -1.0e37f) ? __expf(sv[g][r] - newm[r]) : 0.f;
        p[g][r] = pv; rs[r] += pv;
      }
#pragma unroll
    for (int off = 1; off < 16; off <<= 1)
#pragma unroll
      for (int r = 0; r < 8; ++r) rs[r] += __shfl_xor(rs[r], off, 32);
#pragma unroll
    for (int r = 0; r < 8; ++r) l_r[r] = l_r[r] * alpha[r] + rs[r];
#pragma unroll
    for (int n = 0; n < 4; ++n)
#pragma unroll
      for (int r = 0; r < 8; ++r) acc[n][r] *= alpha[r];

    // ---- restripe P (C-layout -> A-layout) through LDS ----
#pragma unroll
    for (int g = 0; g < 2; ++g)
#pragma unroll
      for (int r = 0; r < 8; ++r)
        Plds[r + half * 8][g * 16 + lh] = (_Float16)p[g][r];
    __syncthreads();
    v16h ap;
#pragma unroll
    for (int e = 0; e < 16; ++e) {
      int K = half * 16 + e;
      ap[e] = Plds[lh][K];
    }

    // ---- O += P V  (4 output column tiles of 16) ----
#pragma unroll
    for (int n = 0; n < 4; ++n) {
      v16h bv;
#pragma unroll
      for (int e = 0; e < 16; ++e) {
        int ktl = half * 16 + e;
        bv[e] = (_Float16)KV[ktl * 128 + 64 + n * 16 + lh];
      }
      acc[n] = wmma_f16(ap, bv, acc[n]);
    }
    __syncthreads();   // protect KV/Plds before next iteration's DMA/stores
  }

#pragma unroll
  for (int n = 0; n < 4; ++n)
#pragma unroll
    for (int r = 0; r < 8; ++r) {
      int qt = r0 + r + half * 8;
      out[(size_t)qt * HD + h * DH + n * 16 + lh] = acc[n][r] / l_r[r];
    }
}

// ---------------------------------------------------------------------------
// Compression pipeline helpers
// ---------------------------------------------------------------------------
__device__ __forceinline__ unsigned hash_u32(unsigned x) {
  x ^= x >> 16; x *= 0x7feb352dU; x ^= x >> 15; x *= 0x846ca68bU; x ^= x >> 16;
  return x;
}
__device__ __forceinline__ float gaussf(int seed, int idx) {
  unsigned h1 = hash_u32((unsigned)(idx * 2 + 1) ^ ((unsigned)seed * 0x9e3779b9U));
  unsigned h2 = hash_u32(h1 + 0x9e3779b9U);
  float u1 = (float)(h1 >> 8) * (1.0f / 16777216.0f) + 1.0e-7f;
  float u2 = (float)(h2 >> 8) * (1.0f / 16777216.0f);
  return sqrtf(-2.f * __logf(u1)) * __cosf(6.2831853f * u2);
}
__device__ __forceinline__ float resid_at(const float* src, int rs, float thr, int t, int c) {
  float x = src[(size_t)t * rs + c];
  return (fabsf(x) >= thr) ? 0.f : x;
}
__device__ __forceinline__ float lowrank_at(const float* Qm, const float* Rm, int t, int c) {
  float s = 0.f;
#pragma unroll
  for (int r = 0; r < RANK; ++r) s += Qm[(size_t)t * RANK + r] * Rm[(size_t)r * HD + c];
  return s;
}

__global__ void init_kernel(unsigned* mx, unsigned* hist) {
  int i = blockIdx.x * blockDim.x + threadIdx.x;
  if (i < 1024) hist[i] = 0u;
  if (i == 0) { mx[0] = 0u; mx[1] = 0u; }
}

__global__ __launch_bounds__(256)
void absmax_kernel(const float* __restrict__ src, int rs, unsigned* mx) {
  int tot = gridDim.x * blockDim.x;
  float local = 0.f;
  for (int i = blockIdx.x * blockDim.x + threadIdx.x; i < NELEM; i += tot) {
    int t = i / HD, c = i - t * HD;
    local = fmaxf(local, fabsf(src[(size_t)t * rs + c]));
  }
  atomicMax(mx, __float_as_uint(local));   // non-negative floats order as uints
}

__global__ __launch_bounds__(256)
void hist_kernel(const float* __restrict__ src, int rs,
                 const unsigned* __restrict__ mx, unsigned* __restrict__ hist) {
  __shared__ unsigned hloc[1024];
  for (int i = threadIdx.x; i < 1024; i += blockDim.x) hloc[i] = 0u;
  __syncthreads();
  float amax = __uint_as_float(mx[0]);
  float inv = (amax > 0.f) ? (1024.f / amax) : 0.f;
  int tot = gridDim.x * blockDim.x;
  for (int i = blockIdx.x * blockDim.x + threadIdx.x; i < NELEM; i += tot) {
    int t = i / HD, c = i - t * HD;
    int b = (int)(fabsf(src[(size_t)t * rs + c]) * inv);
    if (b > 1023) b = 1023;
    atomicAdd(&hloc[b], 1u);
  }
  __syncthreads();
  for (int i = threadIdx.x; i < 1024; i += blockDim.x)
    if (hloc[i]) atomicAdd(&hist[i], hloc[i]);
}

__global__ void thresh_kernel(const unsigned* mx, const unsigned* hist, float* thr) {
  unsigned target = (unsigned)(NELEM / 100);   // top 1%
  unsigned cum = 0; int b = 1023;
  for (; b > 0; --b) { cum += hist[b]; if (cum >= target) break; }
  *thr = __uint_as_float(mx[0]) * ((float)b / 1024.f);
}

__global__ __launch_bounds__(256)
void split_kernel(const float* __restrict__ src, int rs,
                  const float* __restrict__ thrp, float* __restrict__ dst) {
  float thr = *thrp;
  int tot = gridDim.x * blockDim.x;
  for (int i = blockIdx.x * blockDim.x + threadIdx.x; i < NELEM; i += tot) {
    int t = i / HD, c = i - t * HD;
    float x = src[(size_t)t * rs + c];
    dst[i] = (fabsf(x) >= thr) ? x : 0.f;
  }
}

// ---------------------------------------------------------------------------
// Skinny f32 matmuls via V_WMMA_F32_16X16X4_F32 (modes documented in round 1)
// ---------------------------------------------------------------------------
__global__ __launch_bounds__(32)
void mm_kernel(int mode, const float* __restrict__ src, int rowstride,
               const float* __restrict__ thrp, const float* __restrict__ Bmat,
               int seed, int Kdim, float* __restrict__ outp) {
  const int lane = threadIdx.x, lh = lane & 15, half = lane >> 4;
  const int tile = blockIdx.x;
  const float thr = *thrp;
  v8f c = {};
  for (int k0 = 0; k0 < Kdim; k0 += 4) {
    const int ka = k0 + half * 2;
    v2f a, b;
    if (mode == 0) {
      int row = tile * 16 + lh;
      a.x = resid_at(src, rowstride, thr, row, ka);
      a.y = resid_at(src, rowstride, thr, row, ka + 1);
      b.x = gaussf(seed, ka * RANK + lh);
      b.y = gaussf(seed, (ka + 1) * RANK + lh);
    } else if (mode == 1) {
      int row = tile * 16 + lh;
      a.x = resid_at(src, rowstride, thr, row, ka);
      a.y = resid_at(src, rowstride, thr, row, ka + 1);
      b.x = Bmat[(size_t)ka * RANK + lh];
      b.y = Bmat[(size_t)(ka + 1) * RANK + lh];
    } else if (mode == 2) {
      int row = tile * 16 + lh;                 // column index j of resid
      a.x = resid_at(src, rowstride, thr, ka,     row);
      a.y = resid_at(src, rowstride, thr, ka + 1, row);
      b.x = Bmat[(size_t)ka * RANK + lh];
      b.y = Bmat[(size_t)(ka + 1) * RANK + lh];
    } else {
      a.x = Bmat[(size_t)ka * RANK + lh];       // A(r=lh, t=ka) = Q[t][r]
      a.y = Bmat[(size_t)(ka + 1) * RANK + lh];
      int col = tile * 16 + lh;
      b.x = resid_at(src, rowstride, thr, ka,     col);
      b.y = resid_at(src, rowstride, thr, ka + 1, col);
    }
    c = wmma_f32x4(a, b, c);
  }
  if (mode <= 2) {
#pragma unroll
    for (int r = 0; r < 8; ++r)
      outp[(size_t)(tile * 16 + r + half * 8) * RANK + lh] = c[r];
  } else {
#pragma unroll
    for (int r = 0; r < 8; ++r)
      outp[(size_t)(r + half * 8) * HD + tile * 16 + lh] = c[r];
  }
}

__global__ __launch_bounds__(256)
void gram_kernel(const float* __restrict__ Y, int M, float* __restrict__ gram) {
  int a = threadIdx.x & 15, b = threadIdx.x >> 4;
  float s = 0.f;
  for (int t = 0; t < M; ++t) s += Y[(size_t)t * RANK + a] * Y[(size_t)t * RANK + b];
  gram[b * RANK + a] = s;
}

__global__ void chol_kernel(const float* __restrict__ g, float* __restrict__ R) {
  for (int i = 0; i < RANK; ++i) {
    float d = g[i * RANK + i];
    for (int k = 0; k < i; ++k) d -= R[k * RANK + i] * R[k * RANK + i];
    d = sqrtf(fmaxf(d, 1e-20f));
    R[i * RANK + i] = d;
    for (int j = i + 1; j < RANK; ++j) {
      float v = g[i * RANK + j];
      for (int k = 0; k < i; ++k) v -= R[k * RANK + i] * R[k * RANK + j];
      R[i * RANK + j] = v / d;
    }
    for (int j = 0; j < i; ++j) R[i * RANK + j] = 0.f;
  }
}

__global__ __launch_bounds__(256)
void rinv_kernel(const float* __restrict__ Y, int M,
                 const float* __restrict__ R, float* __restrict__ Q) {
  int row = blockIdx.x * blockDim.x + threadIdx.x;
  if (row >= M) return;
  float q[RANK];
#pragma unroll
  for (int c = 0; c < RANK; ++c) {
    float v = Y[(size_t)row * RANK + c];
    for (int i = 0; i < c; ++i) v -= q[i] * R[i * RANK + c];
    q[c] = v / R[c * RANK + c];
  }
#pragma unroll
  for (int c = 0; c < RANK; ++c) Q[(size_t)row * RANK + c] = q[c];
}

__global__ __launch_bounds__(256)
void r2max_kernel(const float* __restrict__ src, int rs, const float* __restrict__ thrp,
                  const float* __restrict__ Qm, const float* __restrict__ Rm,
                  unsigned* mx) {
  float thr = *thrp;
  int tot = gridDim.x * blockDim.x;
  float local = 0.f;
  for (int i = blockIdx.x * blockDim.x + threadIdx.x; i < NELEM; i += tot) {
    int t = i / HD, c = i - t * HD;
    float r2 = resid_at(src, rs, thr, t, c) - lowrank_at(Qm, Rm, t, c);
    local = fmaxf(local, fabsf(r2));
  }
  atomicMax(mx, __float_as_uint(local));
}

__global__ __launch_bounds__(256)
void final_kernel(const float* __restrict__ src, int rs, const float* __restrict__ thrp,
                  const float* __restrict__ Qm, const float* __restrict__ Rm,
                  const unsigned* __restrict__ mxr2, float* __restrict__ dst) {
  float thr = *thrp;
  float scale = __uint_as_float(*mxr2) / 127.f;   // qmax = 2^(8-1)-1
  int tot = gridDim.x * blockDim.x;
  for (int i = blockIdx.x * blockDim.x + threadIdx.x; i < NELEM; i += tot) {
    int t = i / HD, c = i - t * HD;
    float low = lowrank_at(Qm, Rm, t, c);
    float r2 = resid_at(src, rs, thr, t, c) - low;
    float deq = 0.f;
    if (scale > 0.f) {
      float qv = fminf(fmaxf(rintf(r2 / scale), -128.f), 127.f);
      deq = qv * scale;
    }
    dst[i] += low + deq;   // dst already holds outliers from split_kernel
  }
}

// ---------------------------------------------------------------------------
extern "C" void kernel_launch(void* const* d_in, const int* in_sizes, int n_in,
                              void* d_out, int out_size, void* d_ws, size_t ws_size,
                              hipStream_t stream) {
  (void)in_sizes; (void)n_in; (void)out_size; (void)ws_size;
  const float* qkv = (const float*)d_in[0];
  const int*   cu  = (const int*)d_in[1];
  float* out = (float*)d_out;

  // scratch layout (~630 KB)
  unsigned* mx   = (unsigned*)d_ws;            // [0]=absmax, [1]=r2max
  unsigned* hist = mx + 2;                     // 1024 bins
  float* thr  = (float*)(hist + 1024);         // 1
  float* gram = thr + 1;                       // 256
  float* chol = gram + 256;                    // 256
  float* Z    = chol + 256;                    // 768*16
  float* Rm   = Z + (size_t)HD * RANK;         // 16*768
  float* Y    = Rm + (size_t)RANK * HD;        // 4096*16
  float* Qm   = Y + (size_t)T_TOK * RANK;      // 4096*16

  // 1) attention output -> d_out[0 : T*H*D)
  fa_varlen_kernel<<<dim3(T_TOK / 16, NH), 32, 0, stream>>>(qkv, cu, out);

  // 2) compress/decompress roundtrip for q, k, v, out
  for (int i = 0; i < 4; ++i) {
    const float* src; int rs;
    if (i < 3) { src = qkv + i * HD; rs = QKV_STRIDE; }
    else       { src = out;          rs = HD; }
    float* dst = out + (size_t)(i + 1) * NELEM;

    init_kernel<<<4, 256, 0, stream>>>(mx, hist);
    absmax_kernel<<<2048, 256, 0, stream>>>(src, rs, mx);
    hist_kernel<<<2048, 256, 0, stream>>>(src, rs, mx, hist);
    thresh_kernel<<<1, 1, 0, stream>>>(mx, hist, thr);
    split_kernel<<<2048, 256, 0, stream>>>(src, rs, thr, dst);

    // Q1 = qr(resid @ G)
    mm_kernel<<<T_TOK / 16, 32, 0, stream>>>(0, src, rs, thr, nullptr, i, HD, Y);
    gram_kernel<<<1, 256, 0, stream>>>(Y, T_TOK, gram);
    chol_kernel<<<1, 1, 0, stream>>>(gram, chol);
    rinv_kernel<<<T_TOK / 256, 256, 0, stream>>>(Y, T_TOK, chol, Qm);

    // Z = qr(resid^T @ Q1)
    mm_kernel<<<HD / 16, 32, 0, stream>>>(2, src, rs, thr, Qm, i, T_TOK, Y);
    gram_kernel<<<1, 256, 0, stream>>>(Y, HD, gram);
    chol_kernel<<<1, 1, 0, stream>>>(gram, chol);
    rinv_kernel<<<HD / 256, 256, 0, stream>>>(Y, HD, chol, Z);

    // Q = qr(resid @ Z)
    mm_kernel<<<T_TOK / 16, 32, 0, stream>>>(1, src, rs, thr, Z, i, HD, Y);
    gram_kernel<<<1, 256, 0, stream>>>(Y, T_TOK, gram);
    chol_kernel<<<1, 1, 0, stream>>>(gram, chol);
    rinv_kernel<<<T_TOK / 256, 256, 0, stream>>>(Y, T_TOK, chol, Qm);

    // R = Q^T resid
    mm_kernel<<<HD / 16, 32, 0, stream>>>(3, src, rs, thr, Qm, i, T_TOK, Rm);

    // int8 residual quantization scale + reconstruction
    r2max_kernel<<<2048, 256, 0, stream>>>(src, rs, thr, Qm, Rm, mx + 1);
    final_kernel<<<2048, 256, 0, stream>>>(src, rs, thr, Qm, Rm, mx + 1, dst);
  }
}
